// WindowAttention_86045374808609
// MI455X (gfx1250) — matrix operand order repaired and checked by reading, not compiled, not run on hardware
//
#include <hip/hip_runtime.h>
#include <hip/hip_bf16.h>

// ---------------------------------------------------------------------------
// Window attention for MI455X (gfx1250), bf16 WMMA path.
//   B_=1024, N=64, C=512, H=16, d=32, NW=64
// Scratch (bf16): q | k | v | attn_out, each 33,554,432 elems -> 256 MB total.
// ---------------------------------------------------------------------------

typedef __attribute__((ext_vector_type(16))) __bf16 bf16x16;
typedef __attribute__((ext_vector_type(8)))  __bf16 bf16x8;
typedef __attribute__((ext_vector_type(4)))  __bf16 bf16x4;
typedef __attribute__((ext_vector_type(8)))  float  f32x8;
typedef __attribute__((ext_vector_type(4)))  float  f32x4;
typedef __attribute__((ext_vector_type(4)))  int    i32x4;

#define WMMA_BF16(A, B, Cacc) \
  __builtin_amdgcn_wmma_f32_16x16x32_bf16(false, (A), false, (B), (short)0, (Cacc), false, false)

// ---- async global->LDS (CDNA5), guarded so both toolchains compile --------
#if defined(__has_builtin)
#if __has_builtin(__builtin_amdgcn_global_load_async_to_lds_b128) && \
    __has_builtin(__builtin_amdgcn_s_wait_asynccnt)
#define HAVE_ASYNC_LDS 1
#endif
#endif
#ifndef HAVE_ASYNC_LDS
#define HAVE_ASYNC_LDS 0
#endif

typedef __attribute__((address_space(3))) i32x4* lds128_t;

#define ASYNC_B128(gsrc, ldst) \
  __builtin_amdgcn_global_load_async_to_lds_b128((i32x4*)(gsrc), (lds128_t)(ldst), 0, 0)

union frag_u {
  bf16x16 v;
  struct { bf16x8 lo, hi; } h;
};

// A fragment: 16x32 bf16 tile, row-major, leading dim `lda` (16B-aligned rows).
// lane L: row = L&15; half g = L>>4 holds K = g*16 + 0..15 (contiguous).
__device__ __forceinline__ bf16x16 load_frag_a(const __bf16* src, int lda) {
  const int lane = threadIdx.x & 31;
  const int r = lane & 15, g = lane >> 4;
  const __bf16* row = src + r * lda;
  frag_u u;
  u.h.lo = *(const bf16x8*)(row + g * 16);
  u.h.hi = *(const bf16x8*)(row + g * 16 + 8);
  return u.v;
}

// B fragment from TRANSPOSED tile Bt[n][k] (k contiguous, ld = 32).
// lane L: col n = L&15; half g = L>>4 holds K = g*16 + e  -> 32B contiguous.
__device__ __forceinline__ bf16x16 load_frag_bt(const __bf16* src) {
  const int lane = threadIdx.x & 31;
  const int n = lane & 15, g = lane >> 4;
  const __bf16* p = src + n * 32 + g * 16;
  frag_u u;
  u.h.lo = *(const bf16x8*)p;
  u.h.hi = *(const bf16x8*)(p + 8);
  return u.v;
}

// ---------------------------------------------------------------------------
// Kernel 1: qkv = x @ qkv_w + qkv_b ; scatter into q/k/v head tiles (bf16).
// ---------------------------------------------------------------------------
__global__ __launch_bounds__(128)
void qkv_gemm_kernel(const float* __restrict__ x, const float* __restrict__ w,
                     const float* __restrict__ bias,
                     __bf16* __restrict__ qb, __bf16* __restrict__ kb,
                     __bf16* __restrict__ vb) {
  __shared__ __align__(16) __bf16 As[64 * 32];  // [row][k]
  __shared__ __align__(16) __bf16 Bs[64 * 32];  // TRANSPOSED: [col][k]
  const int tid  = threadIdx.x;
  const int lane = tid & 31;
  const int wave = tid >> 5;
  const int wr = wave >> 1, wc = wave & 1;
  const int nl = lane & 15, g = lane >> 4;
  const int row0 = blockIdx.y * 64;
  const int col0 = blockIdx.x * 64;

  f32x8 acc[2][2] = {};

  for (int kt = 0; kt < 16; ++kt) {
#pragma unroll
    for (int i = 0; i < 4; ++i) {             // stage A: 512 float4 chunks
      int c4 = i * 128 + tid;
      int r = c4 >> 3, sub = (c4 & 7) * 4;
      f32x4 fv = *(const f32x4*)&x[(size_t)(row0 + r) * 512 + kt * 32 + sub];
      bf16x4 bv;
#pragma unroll
      for (int j = 0; j < 4; ++j) bv[j] = (__bf16)fv[j];
      *(bf16x4*)&As[r * 32 + sub] = bv;
    }
#pragma unroll
    for (int i = 0; i < 4; ++i) {             // stage B transposed
      int c4 = i * 128 + tid;
      int r = c4 >> 4, cs = (c4 & 15) * 4;    // r = k row, cs = col
      f32x4 fv = *(const f32x4*)&w[(size_t)(kt * 32 + r) * 1536 + col0 + cs];
#pragma unroll
      for (int j = 0; j < 4; ++j) Bs[(cs + j) * 32 + r] = (__bf16)fv[j];
    }
    if (kt + 1 < 16) {                        // global_prefetch_b8 of next tiles
      __builtin_prefetch(&x[(size_t)(row0 + (tid >> 2)) * 512 + (kt + 1) * 32], 0, 1);
      __builtin_prefetch(&w[(size_t)((kt + 1) * 32 + (tid & 31)) * 1536 + col0], 0, 1);
    }
    __syncthreads();

    bf16x16 af[2], bf_[2];
#pragma unroll
    for (int fm = 0; fm < 2; ++fm)
      af[fm] = load_frag_a(&As[(wr * 32 + fm * 16) * 32], 32);
#pragma unroll
    for (int fn = 0; fn < 2; ++fn)
      bf_[fn] = load_frag_bt(&Bs[(wc * 32 + fn * 16) * 32]);
#pragma unroll
    for (int fm = 0; fm < 2; ++fm)
#pragma unroll
      for (int fn = 0; fn < 2; ++fn)
        acc[fm][fn] = WMMA_BF16(af[fm], bf_[fn], acc[fm][fn]);
    __syncthreads();
  }

  const float qscale = 0.17677669529663687f;  // 32^-0.5
#pragma unroll
  for (int fm = 0; fm < 2; ++fm)
#pragma unroll
    for (int fn = 0; fn < 2; ++fn)
#pragma unroll
      for (int j = 0; j < 8; ++j) {
        int row = row0 + wr * 32 + fm * 16 + g * 8 + j;
        int c3  = col0 + wc * 32 + fn * 16 + nl;
        float v = acc[fm][fn][j] + bias[c3];
        int s = c3 >> 9, rem = c3 & 511;
        int h = rem >> 5, dd = rem & 31;
        int b = row >> 6, n = row & 63;
        size_t off = (size_t)(b * 16 + h) * 2048 + n * 32 + dd;
        if (s == 0)      qb[off] = (__bf16)(v * qscale);
        else if (s == 1) kb[off] = (__bf16)v;
        else             vb[off] = (__bf16)v;
      }
}

// ---------------------------------------------------------------------------
// Kernel 2: per (window b, head h): S = Q K^T + bias + mask; P = softmax(S);
// O = P V. One wave per block. Q/K/V staged to LDS via async b128 loads.
// ---------------------------------------------------------------------------
__global__ __launch_bounds__(32)
void attn_kernel(const __bf16* __restrict__ q, const __bf16* __restrict__ k,
                 const __bf16* __restrict__ v, const float* __restrict__ mask,
                 const float* __restrict__ bias_table,
                 __bf16* __restrict__ out) {
  __shared__ __align__(16) __bf16 Qs[64 * 32];
  __shared__ __align__(16) __bf16 Ks[64 * 32];
  __shared__ __align__(16) __bf16 Vs[64 * 32];
  __shared__ __align__(16) float  S[64 * 72];
  __shared__ __align__(16) __bf16 P[64 * 72];
  const int bh = blockIdx.x;
  const int b = bh >> 4, h = bh & 15;
  const int lane = threadIdx.x & 31;
  const int nl = lane & 15, g = lane >> 4;
  const __bf16* qg = q + (size_t)bh * 2048;
  const __bf16* kg = k + (size_t)bh * 2048;
  const __bf16* vg = v + (size_t)bh * 2048;

  // ---- stage Q/K/V tiles (4 KB each) into LDS ----
#if HAVE_ASYNC_LDS
#pragma unroll
  for (int i = 0; i < 8; ++i) {
    int off = (i * 32 + lane) * 8;  // bf16 elements, 16B chunks
    ASYNC_B128(qg + off, Qs + off);
    ASYNC_B128(kg + off, Ks + off);
    ASYNC_B128(vg + off, Vs + off);
  }
  __builtin_amdgcn_s_wait_asynccnt(0);
#else
#pragma unroll
  for (int i = 0; i < 8; ++i) {
    int off = (i * 32 + lane) * 8;
    *(bf16x8*)&Qs[off] = *(const bf16x8*)(qg + off);
    *(bf16x8*)&Ks[off] = *(const bf16x8*)(kg + off);
    *(bf16x8*)&Vs[off] = *(const bf16x8*)(vg + off);
  }
#endif
  __syncthreads();

  // ---- S = Q K^T : A = Q tiles, B = K^T tiles (K is [m][d] => frag-contig) --
  bf16x16 aq[4], bk[4];
#pragma unroll
  for (int mt = 0; mt < 4; ++mt) aq[mt] = load_frag_a(&Qs[mt * 16 * 32], 32);
#pragma unroll
  for (int nt = 0; nt < 4; ++nt) bk[nt] = load_frag_bt(&Ks[nt * 16 * 32]);
  f32x8 sacc[4][4] = {};
#pragma unroll
  for (int mt = 0; mt < 4; ++mt)
#pragma unroll
    for (int nt = 0; nt < 4; ++nt)
      sacc[mt][nt] = WMMA_BF16(aq[mt], bk[nt], sacc[mt][nt]);

  // ---- add relative-position bias + window mask, spill to LDS ----
  const int wdx = b & 63;
#pragma unroll
  for (int mt = 0; mt < 4; ++mt)
#pragma unroll
    for (int nt = 0; nt < 4; ++nt)
#pragma unroll
      for (int j = 0; j < 8; ++j) {
        int n = mt * 16 + g * 8 + j;
        int m = nt * 16 + nl;
        S[n * 72 + m] = sacc[mt][nt][j] + bias_table[(n - m + 63) * 16 + h] +
                        mask[(size_t)wdx * 4096 + n * 64 + m];
      }
  __syncthreads();

  // ---- softmax: 2 rows per lane ----
#pragma unroll
  for (int rr = 0; rr < 2; ++rr) {
    int r = lane * 2 + rr;
    float mx = -1e30f;
    for (int c = 0; c < 64; ++c) mx = fmaxf(mx, S[r * 72 + c]);
    float sum = 0.f;
    for (int c = 0; c < 64; ++c) {
      float e = __expf(S[r * 72 + c] - mx);
      sum += e;
      S[r * 72 + c] = e;
    }
    float inv = 1.f / sum;
    for (int c = 0; c < 64; ++c) P[r * 72 + c] = (__bf16)(S[r * 72 + c] * inv);
  }
  __syncthreads();

  // ---- O = P V ----
  f32x8 o[4][2] = {};
#pragma unroll
  for (int ks = 0; ks < 2; ++ks) {
    bf16x16 pa[4], vbf[2];
#pragma unroll
    for (int mt = 0; mt < 4; ++mt)
      pa[mt] = load_frag_a(&P[mt * 16 * 72 + ks * 32], 72);
#pragma unroll
    for (int nt = 0; nt < 2; ++nt) {
      bf16x16 f;
#pragma unroll
      for (int e = 0; e < 16; ++e)
        f[e] = Vs[(ks * 32 + g * 16 + e) * 32 + nt * 16 + nl];
      vbf[nt] = f;
    }
#pragma unroll
    for (int mt = 0; mt < 4; ++mt)
#pragma unroll
      for (int nt = 0; nt < 2; ++nt)
        o[mt][nt] = WMMA_BF16(pa[mt], vbf[nt], o[mt][nt]);
  }

  // ---- store per reference's [H,B_,N,d] flatten: tile at h*2M + b*2048 ----
  __bf16* ob = out + (size_t)h * (1024 * 2048) + (size_t)b * 2048;
#pragma unroll
  for (int mt = 0; mt < 4; ++mt)
#pragma unroll
    for (int nt = 0; nt < 2; ++nt)
#pragma unroll
      for (int j = 0; j < 8; ++j) {
        int n = mt * 16 + g * 8 + j;
        int dd = nt * 16 + nl;
        ob[n * 32 + dd] = (__bf16)o[mt][nt][j];
      }
}

// ---------------------------------------------------------------------------
// Kernel 3: out = A @ proj_w + proj_b. A: [65536,512] bf16, w: [512,512] f32.
// A tiles staged with async global->LDS b128; B transposed in LDS.
// ---------------------------------------------------------------------------
__global__ __launch_bounds__(128)
void proj_gemm_kernel(const __bf16* __restrict__ A, const float* __restrict__ w,
                      const float* __restrict__ bias, float* __restrict__ out) {
  __shared__ __align__(16) __bf16 As[64 * 32];  // [row][k]
  __shared__ __align__(16) __bf16 Bs[64 * 32];  // TRANSPOSED: [col][k]
  const int tid  = threadIdx.x;
  const int lane = tid & 31;
  const int wave = tid >> 5;
  const int wr = wave >> 1, wc = wave & 1;
  const int nl = lane & 15, g = lane >> 4;
  const int row0 = blockIdx.y * 64;
  const int col0 = blockIdx.x * 64;

  f32x8 acc[2][2] = {};

  for (int kt = 0; kt < 16; ++kt) {
#if HAVE_ASYNC_LDS
#pragma unroll
    for (int i = 0; i < 2; ++i) {             // A: 256 chunks of 16B, async
      int c = i * 128 + tid;
      int r = c >> 2, sub = (c & 3) * 8;
      ASYNC_B128(&A[(size_t)(row0 + r) * 512 + kt * 32 + sub], &As[r * 32 + sub]);
    }
#else
#pragma unroll
    for (int i = 0; i < 2; ++i) {
      int c = i * 128 + tid;
      int r = c >> 2, sub = (c & 3) * 8;
      *(bf16x8*)&As[r * 32 + sub] =
          *(const bf16x8*)&A[(size_t)(row0 + r) * 512 + kt * 32 + sub];
    }
#endif
#pragma unroll
    for (int i = 0; i < 4; ++i) {             // stage B transposed, f32->bf16
      int c4 = i * 128 + tid;
      int r = c4 >> 4, cs = (c4 & 15) * 4;
      f32x4 fv = *(const f32x4*)&w[(size_t)(kt * 32 + r) * 512 + col0 + cs];
#pragma unroll
      for (int j = 0; j < 4; ++j) Bs[(cs + j) * 32 + r] = (__bf16)fv[j];
    }
    if (kt + 1 < 16) {
      __builtin_prefetch(&A[(size_t)(row0 + (tid >> 2)) * 512 + (kt + 1) * 32], 0, 1);
      __builtin_prefetch(&w[(size_t)((kt + 1) * 32 + (tid & 31)) * 512 + col0], 0, 1);
    }
#if HAVE_ASYNC_LDS
    __builtin_amdgcn_s_wait_asynccnt(0);
#endif
    __syncthreads();

    bf16x16 af[2], bf_[2];
#pragma unroll
    for (int fm = 0; fm < 2; ++fm)
      af[fm] = load_frag_a(&As[(wr * 32 + fm * 16) * 32], 32);
#pragma unroll
    for (int fn = 0; fn < 2; ++fn)
      bf_[fn] = load_frag_bt(&Bs[(wc * 32 + fn * 16) * 32]);
#pragma unroll
    for (int fm = 0; fm < 2; ++fm)
#pragma unroll
      for (int fn = 0; fn < 2; ++fn)
        acc[fm][fn] = WMMA_BF16(af[fm], bf_[fn], acc[fm][fn]);
    __syncthreads();
  }

#pragma unroll
  for (int fm = 0; fm < 2; ++fm)
#pragma unroll
    for (int fn = 0; fn < 2; ++fn)
#pragma unroll
      for (int j = 0; j < 8; ++j) {
        int row = row0 + wr * 32 + fm * 16 + g * 8 + j;
        int col = col0 + wc * 32 + fn * 16 + nl;
        out[(size_t)row * 512 + col] = acc[fm][fn][j] + bias[col];
      }
}

// ---------------------------------------------------------------------------
extern "C" void kernel_launch(void* const* d_in, const int* in_sizes, int n_in,
                              void* d_out, int out_size, void* d_ws, size_t ws_size,
                              hipStream_t stream) {
  (void)in_sizes; (void)n_in; (void)out_size; (void)ws_size;
  const float* x      = (const float*)d_in[0];  // [1024,64,512]
  const float* mask   = (const float*)d_in[1];  // [64,64,64]
  const float* qkv_w  = (const float*)d_in[2];  // [512,1536]
  const float* qkv_b  = (const float*)d_in[3];  // [1536]
  const float* proj_w = (const float*)d_in[4];  // [512,512]
  const float* proj_b = (const float*)d_in[5];  // [512]
  const float* btab   = (const float*)d_in[6];  // [127,16]
  float* out = (float*)d_out;                   // [1024,64,512] f32

  const size_t HEAD_ELEMS = (size_t)1024 * 16 * 64 * 32;  // 33,554,432
  __bf16* qb = (__bf16*)d_ws;
  __bf16* kb = qb + HEAD_ELEMS;
  __bf16* vb = kb + HEAD_ELEMS;
  __bf16* ab = vb + HEAD_ELEMS;

  qkv_gemm_kernel<<<dim3(1536 / 64, 65536 / 64), 128, 0, stream>>>(
      x, qkv_w, qkv_b, qb, kb, vb);
  attn_kernel<<<dim3(1024 * 16), 32, 0, stream>>>(qb, kb, vb, mask, btab, ab);
  proj_gemm_kernel<<<dim3(512 / 64, 65536 / 64), 128, 0, stream>>>(
      ab, proj_w, proj_b, out);
}
